// MyLLMFFNMoE_55250459295817
// MI455X (gfx1250) — matrix operand order repaired and checked by reading, not compiled, not run on hardware
//
#include <hip/hip_runtime.h>
#include <hip/hip_bf16.h>
#include <stdint.h>

#define T_TOK   8192
#define DMODEL  2048
#define NEXP    6
#define HEXP    1024
#define HSHARE  2048

// LDS row strides (elements), padded so every row is 16B aligned and bank-spread
constexpr int AST = DMODEL + 8;   // 2056 -> 4112 B/row (16B aligned)
constexpr int HST = HEXP + 8;     // 1032 -> 2064 B/row (16B aligned)

typedef __attribute__((ext_vector_type(16))) __bf16 bf16v16;
typedef __attribute__((ext_vector_type(8)))  float  f32v8;

union FragU { unsigned int u[8]; bf16v16 v; };

// Pack two fp32 into packed bf16x2 with ONE v_perm_b32 (truncation rounding):
// result = { hi[31:16], lo[31:16] }  (element 2j in low half, 2j+1 in high half)
__device__ __forceinline__ unsigned int pkbf(float lo, float hi) {
  return __builtin_amdgcn_perm(__float_as_uint(hi), __float_as_uint(lo),
                               0x07060302u);
}
__device__ __forceinline__ unsigned short bf1(float f) {
  return (unsigned short)(__float_as_uint(f) >> 16);   // truncate to bf16
}
// silu via v_rcp_f32 instead of IEEE divide ladder
__device__ __forceinline__ float silu_f(float g) {
  return g * __builtin_amdgcn_rcpf(1.0f + __expf(-g));
}

// ---------------------------------------------------------------------------
// Kernel 1: router. One wave32 per token: gate = x@Wg + bg, keep top-6 of 8
// (drop 2 smallest, jax top_k tie-break = lower index wins), masked softmax,
// write w[t][0..5] to workspace.
// ---------------------------------------------------------------------------
__global__ __launch_bounds__(256)
void moe_router(const float* __restrict__ x, const float* __restrict__ Wg,
                const float* __restrict__ bg, float* __restrict__ wout) {
  const int lane = threadIdx.x & 31;
  const int wav  = threadIdx.x >> 5;
  const int t    = blockIdx.x * 8 + wav;

  float acc[8] = {0.f,0.f,0.f,0.f,0.f,0.f,0.f,0.f};
  const float* xr = x + (size_t)t * DMODEL;
  for (int k = lane; k < DMODEL; k += 32) {
    const float xv = xr[k];
    const float4 w0 = *(const float4*)(Wg + (size_t)k * 8);
    const float4 w1 = *(const float4*)(Wg + (size_t)k * 8 + 4);
    acc[0] += xv * w0.x; acc[1] += xv * w0.y; acc[2] += xv * w0.z; acc[3] += xv * w0.w;
    acc[4] += xv * w1.x; acc[5] += xv * w1.y; acc[6] += xv * w1.z; acc[7] += xv * w1.w;
  }
#pragma unroll
  for (int e = 0; e < 8; ++e)
#pragma unroll
    for (int off = 16; off > 0; off >>= 1)
      acc[e] += __shfl_xor(acc[e], off, 32);   // wave32 butterfly reduce

  float g[8];
#pragma unroll
  for (int e = 0; e < 8; ++e) g[e] = acc[e] + bg[e];

  bool keep[8];
  float m = -3.0e38f;
#pragma unroll
  for (int j = 0; j < 8; ++j) {
    int rank = 0;
#pragma unroll
    for (int k = 0; k < 8; ++k)
      rank += (g[k] > g[j] || (g[k] == g[j] && k < j)) ? 1 : 0;
    keep[j] = (rank < 6);
    if (keep[j] && g[j] > m) m = g[j];
  }
  float Z = 0.f;
#pragma unroll
  for (int j = 0; j < 8; ++j) if (keep[j]) Z += __expf(g[j] - m);
  if (lane < NEXP)
    wout[(size_t)t * NEXP + lane] =
        keep[lane] ? __expf(g[lane] - m) * __builtin_amdgcn_rcpf(Z) : 0.f;
}

// Stage a 32 x DMODEL fp32 tile into LDS as bf16 (packed dword writes).
__device__ __forceinline__ void stage_x32(const float* __restrict__ x, int t0,
                                          unsigned short* __restrict__ Abuf,
                                          int tid) {
  for (int i = tid; i < 32 * (DMODEL / 4); i += 256) {
    const int row = i / (DMODEL / 4), c4 = (i % (DMODEL / 4)) * 4;
    const float4 v = *(const float4*)(x + (size_t)(t0 + row) * DMODEL + c4);
    unsigned int* d = (unsigned int*)(Abuf + row * AST + c4);
    d[0] = pkbf(v.x, v.y);
    d[1] = pkbf(v.z, v.w);
  }
}

// Load the 16x32 bf16 A fragment (ISA layout) from an LDS row-major tile.
// row = matrix row for this lane (pass arow or 16+arow for the second tile).
__device__ __forceinline__ FragU lds_afrag(const unsigned short* __restrict__ buf,
                                           int stride, int row, int akb, int k0) {
  FragU a;
  const unsigned int* a0 = (const unsigned int*)(buf + row * stride + k0 + akb);
  const unsigned int* a1 = (const unsigned int*)(buf + row * stride + k0 + 16 + akb);
#pragma unroll
  for (int j = 0; j < 4; ++j) { a.u[j] = a0[j]; a.u[4 + j] = a1[j]; }
  return a;
}

// Load a 32x16 bf16 B fragment from a row-major fp32 matrix (ld = row stride).
__device__ __forceinline__ FragU gl_bfrag(const float* __restrict__ p, int ld) {
  FragU b;
#pragma unroll
  for (int j = 0; j < 8; ++j)
    b.u[j] = pkbf(p[(2 * j + (j >= 4 ? 8 : 0)) * ld], p[(2 * j + 1 + (j >= 4 ? 8 : 0)) * ld]);
  return b;
}

#define WMMA_BF16(A, B, C)                                                   \
  __builtin_amdgcn_wmma_f32_16x16x32_bf16(false, (A).v, false, (B).v,        \
                                          (short)0, (C), false, false)

// ---------------------------------------------------------------------------
// Kernel 2: shared experts. out = silu(x@Ws_up + bs_up) @ Ws_down + bs_down.
// Block = 32 tokens (two 16-row A tiles), 8 waves. Each B fragment feeds two
// WMMAs -> half the weight traffic per FLOP. Plain stores initialize out.
// ---------------------------------------------------------------------------
__global__ __launch_bounds__(256)
void moe_shared(const float* __restrict__ x,
                const float* __restrict__ Wsu, const float* __restrict__ bsu,
                const float* __restrict__ Wsd, const float* __restrict__ bsd,
                float* __restrict__ out) {
  __shared__ unsigned short Abuf[32 * AST];   // x tile bf16 (32 x 2048)
  __shared__ unsigned short Hbuf[32 * AST];   // silu intermediate (32 x 2048)

  const int tid  = threadIdx.x;
  const int lane = tid & 31;
  const int wav  = tid >> 5;
  const int t0   = blockIdx.x * 32;

  stage_x32(x, t0, Abuf, tid);
  __syncthreads();

  const int arow = lane & 15, akb = (lane >> 4) * 8;
  const int bcol = lane & 15, bkb = (lane >> 4) * 8;
  const int rbase = (lane >> 4) * 8;

  // phase 1: h = silu(x @ Ws_up + bs_up); wave covers 256 cols of HSHARE
  for (int np = 0; np < 8; ++np) {
    const int col0 = wav * 256 + np * 32 + bcol;
    f32v8 a00 = {}, a01 = {}, a10 = {}, a11 = {};
#pragma unroll 2
    for (int k0 = 0; k0 < DMODEL; k0 += 32) {
      const FragU fa0 = lds_afrag(Abuf, AST, arow, akb, k0);
      const FragU fa1 = lds_afrag(Abuf, AST, 16 + arow, akb, k0);
      const float* pb = Wsu + (size_t)(k0 + bkb) * HSHARE + col0;
      __builtin_prefetch(pb + 32 * HSHARE, 0, 1);
      const FragU b0 = gl_bfrag(pb, HSHARE);
      const FragU b1 = gl_bfrag(pb + 16, HSHARE);
      a00 = WMMA_BF16(fa0, b0, a00);  a01 = WMMA_BF16(fa0, b1, a01);
      a10 = WMMA_BF16(fa1, b0, a10);  a11 = WMMA_BF16(fa1, b1, a11);
    }
    const float bv0 = bsu[col0], bv1 = bsu[col0 + 16];
#pragma unroll
    for (int r = 0; r < 8; ++r) {
      Hbuf[(rbase + r) * AST + col0]           = bf1(silu_f(a00[r] + bv0));
      Hbuf[(rbase + r) * AST + col0 + 16]      = bf1(silu_f(a01[r] + bv1));
      Hbuf[(16 + rbase + r) * AST + col0]      = bf1(silu_f(a10[r] + bv0));
      Hbuf[(16 + rbase + r) * AST + col0 + 16] = bf1(silu_f(a11[r] + bv1));
    }
  }
  __syncthreads();

  // phase 2: out = h @ Ws_down + bs_down
  for (int np = 0; np < 8; ++np) {
    const int col0 = wav * 256 + np * 32 + bcol;
    f32v8 a00 = {}, a01 = {}, a10 = {}, a11 = {};
#pragma unroll 2
    for (int k0 = 0; k0 < HSHARE; k0 += 32) {
      const FragU fa0 = lds_afrag(Hbuf, AST, arow, akb, k0);
      const FragU fa1 = lds_afrag(Hbuf, AST, 16 + arow, akb, k0);
      const float* pb = Wsd + (size_t)(k0 + bkb) * DMODEL + col0;
      __builtin_prefetch(pb + 32 * DMODEL, 0, 1);
      const FragU b0 = gl_bfrag(pb, DMODEL);
      const FragU b1 = gl_bfrag(pb + 16, DMODEL);
      a00 = WMMA_BF16(fa0, b0, a00);  a01 = WMMA_BF16(fa0, b1, a01);
      a10 = WMMA_BF16(fa1, b0, a10);  a11 = WMMA_BF16(fa1, b1, a11);
    }
    const float bv0 = bsd[col0], bv1 = bsd[col0 + 16];
#pragma unroll
    for (int r = 0; r < 8; ++r) {
      const size_t ro0 = (size_t)(t0 + rbase + r) * DMODEL;
      const size_t ro1 = (size_t)(t0 + 16 + rbase + r) * DMODEL;
      out[ro0 + col0]      = a00[r] + bv0;
      out[ro0 + col0 + 16] = a01[r] + bv1;
      out[ro1 + col0]      = a10[r] + bv0;
      out[ro1 + col0 + 16] = a11[r] + bv1;
    }
  }
}

// ---------------------------------------------------------------------------
// Kernel 3: routed experts. Block = 32 tokens; loops all 6 experts.
// Per expert: h = silu(x@Wgate+bg)*(x@Wup+bu) -> LDS; then
// out[t,:] += w[t,e] * (h @ Wdown + bdown) via ordinary rmw (each element is
// owned by exactly one thread; experts are sequential -> deterministic, and
// the 32x2048 out tile stays L2-resident across the 6 expert iterations).
// ---------------------------------------------------------------------------
__global__ __launch_bounds__(256)
void moe_routed(const float* __restrict__ x,
                const float* __restrict__ Wgate, const float* __restrict__ bgate,
                const float* __restrict__ Wup,   const float* __restrict__ bup,
                const float* __restrict__ Wdown, const float* __restrict__ bdown,
                const float* __restrict__ wtok,  // T x 6 router weights
                float* __restrict__ out) {
  __shared__ unsigned short Abuf[32 * AST];   // x tile bf16 (32 x 2048)
  __shared__ unsigned short Hbuf[32 * HST];   // SwiGLU intermediate (32 x 1024)

  const int tid  = threadIdx.x;
  const int lane = tid & 31;
  const int wav  = tid >> 5;
  const int t0   = blockIdx.x * 32;

  stage_x32(x, t0, Abuf, tid);
  __syncthreads();

  const int arow = lane & 15, akb = (lane >> 4) * 8;
  const int bcol = lane & 15, bkb = (lane >> 4) * 8;
  const int rbase = (lane >> 4) * 8;

  for (int e = 0; e < NEXP; ++e) {
    const float* Wg_e = Wgate + (size_t)e * DMODEL * HEXP;
    const float* Wu_e = Wup   + (size_t)e * DMODEL * HEXP;
    const float* Wd_e = Wdown + (size_t)e * HEXP * DMODEL;

    // phase 1: wave covers 128 h-cols; each B fragment feeds 2 tiles,
    // each A fragment feeds gate+up -> 4 WMMAs per (A0,A1,Bg,Bu) group
    for (int nt = 0; nt < 8; ++nt) {
      const int col = wav * 128 + nt * 16 + bcol;
      f32v8 g0 = {}, u0 = {}, g1 = {}, u1 = {};
#pragma unroll 2
      for (int k0 = 0; k0 < DMODEL; k0 += 32) {
        const FragU fa0 = lds_afrag(Abuf, AST, arow, akb, k0);
        const FragU fa1 = lds_afrag(Abuf, AST, 16 + arow, akb, k0);
        const float* pg = Wg_e + (size_t)(k0 + bkb) * HEXP + col;
        const float* pu = Wu_e + (size_t)(k0 + bkb) * HEXP + col;
        __builtin_prefetch(pg + 32 * HEXP, 0, 1);
        __builtin_prefetch(pu + 32 * HEXP, 0, 1);
        const FragU fbg = gl_bfrag(pg, HEXP);
        const FragU fbu = gl_bfrag(pu, HEXP);
        g0 = WMMA_BF16(fa0, fbg, g0);  g1 = WMMA_BF16(fa1, fbg, g1);
        u0 = WMMA_BF16(fa0, fbu, u0);  u1 = WMMA_BF16(fa1, fbu, u1);
      }
      const float bgv = bgate[(size_t)e * HEXP + col];
      const float buv = bup[(size_t)e * HEXP + col];
#pragma unroll
      for (int r = 0; r < 8; ++r) {
        Hbuf[(rbase + r) * HST + col]      = bf1(silu_f(g0[r] + bgv) * (u0[r] + buv));
        Hbuf[(16 + rbase + r) * HST + col] = bf1(silu_f(g1[r] + bgv) * (u1[r] + buv));
      }
    }
    __syncthreads();   // Hbuf complete before any wave reads full K

    // per-lane router weights for this expert (both row tiles)
    float w0[8], w1[8];
#pragma unroll
    for (int r = 0; r < 8; ++r) {
      w0[r] = wtok[(size_t)(t0 + rbase + r) * NEXP + e];
      w1[r] = wtok[(size_t)(t0 + 16 + rbase + r) * NEXP + e];
    }

    // phase 2: out += w * (h @ Wdown + bdown); wave covers 256 out cols
    for (int np = 0; np < 8; ++np) {
      const int col0 = wav * 256 + np * 32 + bcol;
      f32v8 a00 = {}, a01 = {}, a10 = {}, a11 = {};
#pragma unroll 2
      for (int k0 = 0; k0 < HEXP; k0 += 32) {
        const FragU fa0 = lds_afrag(Hbuf, HST, arow, akb, k0);
        const FragU fa1 = lds_afrag(Hbuf, HST, 16 + arow, akb, k0);
        const float* pb = Wd_e + (size_t)(k0 + bkb) * DMODEL + col0;
        __builtin_prefetch(pb + 32 * DMODEL, 0, 1);
        const FragU b0 = gl_bfrag(pb, DMODEL);
        const FragU b1 = gl_bfrag(pb + 16, DMODEL);
        a00 = WMMA_BF16(fa0, b0, a00);  a01 = WMMA_BF16(fa0, b1, a01);
        a10 = WMMA_BF16(fa1, b0, a10);  a11 = WMMA_BF16(fa1, b1, a11);
      }
      const float bd0 = bdown[(size_t)e * DMODEL + col0];
      const float bd1 = bdown[(size_t)e * DMODEL + col0 + 16];
#pragma unroll
      for (int r = 0; r < 8; ++r) {
        const size_t ro0 = (size_t)(t0 + rbase + r) * DMODEL;
        const size_t ro1 = (size_t)(t0 + 16 + rbase + r) * DMODEL;
        out[ro0 + col0]      += w0[r] * (a00[r] + bd0);
        out[ro0 + col0 + 16] += w0[r] * (a01[r] + bd1);
        out[ro1 + col0]      += w1[r] * (a10[r] + bd0);
        out[ro1 + col0 + 16] += w1[r] * (a11[r] + bd1);
      }
    }
    __syncthreads();   // protect Hbuf before next expert overwrites it
  }
}

// ---------------------------------------------------------------------------
extern "C" void kernel_launch(void* const* d_in, const int* in_sizes, int n_in,
                              void* d_out, int out_size, void* d_ws, size_t ws_size,
                              hipStream_t stream) {
  const float* x     = (const float*)d_in[0];
  const float* Wg    = (const float*)d_in[1];
  const float* bg    = (const float*)d_in[2];
  const float* Wgate = (const float*)d_in[3];
  const float* bgate = (const float*)d_in[4];
  const float* Wup   = (const float*)d_in[5];
  const float* bup   = (const float*)d_in[6];
  const float* Wdown = (const float*)d_in[7];
  const float* bdown = (const float*)d_in[8];
  const float* Wsu   = (const float*)d_in[9];
  const float* bsu   = (const float*)d_in[10];
  const float* Wsd   = (const float*)d_in[11];
  const float* bsd   = (const float*)d_in[12];
  float* out  = (float*)d_out;
  float* wtok = (float*)d_ws;              // T x 6 router weights (192 KB)

  (void)in_sizes; (void)n_in; (void)out_size; (void)ws_size;

  moe_router<<<T_TOK / 8, 256, 0, stream>>>(x, Wg, bg, wtok);
  moe_shared<<<T_TOK / 32, 256, 0, stream>>>(x, Wsu, bsu, Wsd, bsd, out);
  moe_routed<<<T_TOK / 32, 256, 0, stream>>>(x, Wgate, bgate, Wup, bup,
                                             Wdown, bdown, wtok, out);
}
